// CLIPFNO2d_6760278523959
// MI455X (gfx1250) — matrix operand order repaired and checked by reading, not compiled, not run on hardware
//
#include <hip/hip_runtime.h>
#include <hip/hip_bf16.h>
#include <math.h>

// ---------------------------------------------------------------------------
// CDNA5 (gfx1250) wave32 WMMA types
// ---------------------------------------------------------------------------
typedef __attribute__((ext_vector_type(16))) __bf16 v16bf;
typedef __attribute__((ext_vector_type(8)))  float  v8f;

#define TWO_PI 6.283185307179586f

__device__ inline float apply_act(float x, int act) {
  if (act == 1) return x > 0.f ? x : 0.f;                              // relu
  if (act == 2) return 0.5f * x * (1.0f + erff(x * 0.70710678118f));   // gelu
  if (act == 3) return x / (1.0f + __expf(-x));                        // silu
  return x;
}

// ---------------------------------------------------------------------------
// A-fragment builders (ISA 16-bit A 16x32 layout):
// lane&15 = M row; per lane the needed K values are TWO CONTIGUOUS 8-float
// runs: [k0+khalf .. +7] and [k0+16+khalf .. +7], khalf = (lane>>4)*8.
// ---------------------------------------------------------------------------
__device__ inline v16bf load_a_frag_fast(const float* __restrict__ Ap,
                                         int k0, int khalf) {
  v16bf f;
  const float* p0 = Ap + k0 + khalf;
  const float* p1 = p0 + 16;
#pragma unroll
  for (int e = 0; e < 8; ++e) {
    f[e]     = (__bf16)p0[e];
    f[e + 8] = (__bf16)p1[e];
  }
  return f;
}

// Guarded fallback (edge tiles / single K-tail chunk only)
__device__ inline v16bf load_a_frag_guard(const float* __restrict__ A, long lda,
                                          long row0, long M, int k0, int K,
                                          int lane) {
  v16bf f;
  long row = row0 + (lane & 15);
  bool rok = row < M;
  int khalf = (lane >> 4) * 8;
  const float* Ap = A + row * lda;
#pragma unroll
  for (int e = 0; e < 16; ++e) {
    int k = k0 + ((e < 8) ? 0 : 16) + khalf + (e & 7);
    float v = (rok && k < K) ? Ap[k] : 0.0f;
    f[e] = (__bf16)v;
  }
  return f;
}

// ---------------------------------------------------------------------------
// Generic GEMM: C[M,N] = act(A[M,K] @ B + bias).
// 128 threads = 4 waves; block = 16 rows x 64 cols; one 16x16 C tile / wave.
// B panel staged once into LDS as bf16 [n][k] (pad 16) so each B fragment is
// one aligned 32-byte LDS read shared across waves and k-steps.
// Steady-state loop: 4x global_load_b128 -> cvt_pk_bf16 -> 2x ds_load_b128
// -> v_wmma; the guarded K-tail is peeled out of the loop.
// ---------------------------------------------------------------------------
__global__ void gemm_wmma_kernel(const float* __restrict__ A, long lda,
                                 const float* __restrict__ Bm, long brs, long bcs,
                                 const float* __restrict__ bias,
                                 float* __restrict__ C, long ldc,
                                 long M, long N, int K, int act) {
  extern __shared__ __align__(32) __bf16 ldsB[];
  const int Kp = (K + 31) & ~31;
  const int ldb = Kp + 16;              // elements; byte stride mult of 32
  const int tid = threadIdx.x;
  const long colbase = (long)blockIdx.y * 64;

  // cooperative stage: n = tid&63 fixed per thread (coalesced over cols)
  for (int idx = tid; idx < 64 * Kp; idx += 128) {
    int n = idx & 63, k = idx >> 6;
    long col = colbase + n;
    float v = (col < N && k < K) ? Bm[(long)k * brs + col * bcs] : 0.0f;
    ldsB[n * ldb + ((k & ~31) | (((k >> 3) & 1) << 4) | (((k >> 4) & 1) << 3) | (k & 7))] = (__bf16)v;
  }
  __syncthreads();

  const int lane = tid & 31, wave = tid >> 5;
  const long row0 = (long)blockIdx.x * 16;
  const long col0 = colbase + (long)wave * 16;
  if (row0 < M && col0 < N) {           // wave-uniform
    const float* Ap = A + (row0 + (lane & 15)) * lda;
    const int khalf = (lane >> 4) * 8;
    const int nloc = wave * 16 + (lane & 15);
    const __bf16* bbase = &ldsB[nloc * ldb + (lane >> 4) * 16];

    v8f acc = {};
    if (row0 + 16 <= M) {
      int k0 = 0;
      for (; k0 + 32 <= K; k0 += 32) {       // clean fast loop
        v16bf a = load_a_frag_fast(Ap, k0, khalf);
        v16bf b = *(const v16bf*)(bbase + k0);
        acc = __builtin_amdgcn_wmma_f32_16x16x32_bf16(false, a, false, b, (short)0,
                                                      acc, false, false);
      }
      if (k0 < Kp) {                          // single guarded tail chunk
        v16bf a = load_a_frag_guard(A, lda, row0, M, k0, K, lane);
        v16bf b = *(const v16bf*)(bbase + k0);
        acc = __builtin_amdgcn_wmma_f32_16x16x32_bf16(false, a, false, b, (short)0,
                                                      acc, false, false);
      }
    } else {                                  // rare edge-M tile: all guarded
      for (int k0 = 0; k0 < Kp; k0 += 32) {
        v16bf a = load_a_frag_guard(A, lda, row0, M, k0, K, lane);
        v16bf b = *(const v16bf*)(bbase + k0);
        acc = __builtin_amdgcn_wmma_f32_16x16x32_bf16(false, a, false, b, (short)0,
                                                      acc, false, false);
      }
    }
    long col = col0 + (lane & 15);
    if (col < N) {
      float bv = bias ? bias[col] : 0.0f;
#pragma unroll
      for (int v = 0; v < 8; ++v) {
        long m = row0 + v + ((lane >> 4) * 8);
        if (m < M) C[m * ldc + col] = apply_act(acc[v] + bv, act);
      }
    }
  }
}

// ---------------------------------------------------------------------------
// Acat = concat(x[...,40], grid[...,2]) -> [262144, 42]
// ---------------------------------------------------------------------------
__global__ void concat42_kernel(const float* __restrict__ x,
                                const float* __restrict__ g,
                                float* __restrict__ Acat, long total) {
  long i = (long)blockIdx.x * 256 + threadIdx.x;
  if (i >= total) return;
  long r = i / 42; int k = (int)(i % 42);
  Acat[i] = (k < 40) ? x[r * 40 + k] : g[r * 2 + (k - 40)];
}

// htmp [B,32,32,64] -> h [B,34,34,64] with zero pad on the +2 edges
__global__ void pad_kernel(const float* __restrict__ src, float* __restrict__ h,
                           long total) {
  long i = (long)blockIdx.x * 256 + threadIdx.x;
  if (i >= total) return;
  int c = (int)(i & 63); long t = i >> 6;
  int w = (int)(t % 34); t /= 34;
  int y = (int)(t % 34); long b = t / 34;
  h[i] = (y < 32 && w < 32) ? src[(((b * 32 + y) * 32 + w) << 6) + c] : 0.0f;
}

// ---------------------------------------------------------------------------
// Forward DFT along H at the 24 retained rows.  h -> Ht:[B,24,34,64] complex.
// ---------------------------------------------------------------------------
__global__ void dft_h_fwd_kernel(const float* __restrict__ h,
                                 float* __restrict__ Htr, float* __restrict__ Hti) {
  __shared__ float cs[34], sn[34];
  int bm = blockIdx.x, b = bm / 24, m = bm % 24;
  int r = (m < 12) ? m : (m + 10);
  int tid = threadIdx.x;
  if (tid < 34) {
    float a = TWO_PI * (float)(r * tid) / 34.0f;
    __sincosf(a, &sn[tid], &cs[tid]);
  }
  __syncthreads();
  const float* hb = h + (long)b * 34 * 34 * 64;
  long obase = ((long)b * 24 + m) * 34 * 64;
  for (int idx = tid; idx < 34 * 64; idx += 256) {
    int w = idx >> 6, c = idx & 63;
    float ar = 0.f, ai = 0.f;
    for (int y = 0; y < 34; ++y) {
      float v = hb[((long)y * 34 + w) * 64 + c];
      ar += cs[y] * v;
      ai -= sn[y] * v;
    }
    Htr[obase + idx] = ar;
    Hti[obase + idx] = ai;
  }
}

// Forward DFT along W at the 12 retained cols. Ht -> G:[B,24,12,64] complex.
__global__ void dft_w_fwd_kernel(const float* __restrict__ Htr,
                                 const float* __restrict__ Hti,
                                 float* __restrict__ Gr, float* __restrict__ Gi) {
  __shared__ float cs[12 * 34], sn[12 * 34];
  int bm = blockIdx.x, b = bm / 24, m = bm % 24;
  int tid = threadIdx.x;
  for (int i = tid; i < 12 * 34; i += 256) {
    int n = i / 34, w = i % 34;
    float a = TWO_PI * (float)(n * w) / 34.0f;
    __sincosf(a, &sn[i], &cs[i]);
  }
  __syncthreads();
  long ibase = ((long)b * 24 + m) * 34 * 64;
  long obase = ((long)b * 288 + m * 12) * 64;
  for (int idx = tid; idx < 12 * 64; idx += 256) {
    int n = idx >> 6, c = idx & 63;
    const float* ct = cs + n * 34; const float* st = sn + n * 34;
    float gr = 0.f, gi = 0.f;
    for (int w = 0; w < 34; ++w) {
      float hr = Htr[ibase + (long)w * 64 + c];
      float hi = Hti[ibase + (long)w * 64 + c];
      gr += ct[w] * hr + st[w] * hi;
      gi += ct[w] * hi - st[w] * hr;
    }
    Gr[obase + (long)n * 64 + c] = gr;
    Gi[obase + (long)n * 64 + c] = gi;
  }
}

// ---------------------------------------------------------------------------
// Mode mixing: per mode p, complex GEMM [256x64]@[64x64].  Weight tile staged
// once per block into LDS bf16 (Re/Im planes).  bf16 WMMA forbids A/B NEG so
// Re = ArBr - AiBi uses two accumulators + VALU subtract.
// grid = (16 batch-tiles, 288 modes), 128 threads = 4 waves = 64 output cols.
// ---------------------------------------------------------------------------
__global__ void mode_mix_kernel(const float* __restrict__ Gr,
                                const float* __restrict__ Gi,
                                const float* __restrict__ w1,
                                const float* __restrict__ w2,
                                float* __restrict__ G2r, float* __restrict__ G2i) {
  __shared__ __align__(32) __bf16 wr[64][80];   // [co][ci], pad to 80
  __shared__ __align__(32) __bf16 wi[64][80];
  const int tid = threadIdx.x;
  const int lane = tid & 31, wave = tid >> 5;
  const int p = blockIdx.y;                 // m*12 + n
  const int mm = p / 12, nn = p % 12;
  const float* W = (mm < 12) ? w1 : w2;
  const int mloc = (mm < 12) ? mm : mm - 12;
  const float* W2 = W + (long)(mloc * 12 + nn) * 2;  // (ci,co) at (ci*64+co)*288

  for (int idx = tid; idx < 4096; idx += 128) {
    int ci = idx >> 6, co = idx & 63;
    const float* q = W2 + ((long)ci * 64 + co) * 288;
    int cp = (ci & ~31) | (((ci >> 3) & 1) << 4) | (((ci >> 4) & 1) << 3) | (ci & 7);
    wr[co][cp] = (__bf16)q[0];
    wi[co][cp] = (__bf16)q[1];
  }
  __syncthreads();

  const long row0 = (long)blockIdx.x * 16;
  const long lda = 288L * 64;
  const float* ArB = Gr + (long)p * 64 + (row0 + (lane & 15)) * lda;
  const float* AiB = Gi + (long)p * 64 + (row0 + (lane & 15)) * lda;
  const int khalf = (lane >> 4) * 8;
  const int nloc = wave * 16 + (lane & 15);
  const int koff = (lane >> 4) * 16;

  v8f aRR = {}, aII = {}, aI = {};
#pragma unroll
  for (int k0 = 0; k0 < 64; k0 += 32) {
    v16bf ar = load_a_frag_fast(ArB, k0, khalf);
    v16bf ai = load_a_frag_fast(AiB, k0, khalf);
    v16bf br = *(const v16bf*)&wr[nloc][k0 + koff];
    v16bf bi = *(const v16bf*)&wi[nloc][k0 + koff];
    aRR = __builtin_amdgcn_wmma_f32_16x16x32_bf16(false, ar, false, br, (short)0, aRR, false, false);
    aII = __builtin_amdgcn_wmma_f32_16x16x32_bf16(false, ai, false, bi, (short)0, aII, false, false);
    aI  = __builtin_amdgcn_wmma_f32_16x16x32_bf16(false, ar, false, bi, (short)0, aI,  false, false);
    aI  = __builtin_amdgcn_wmma_f32_16x16x32_bf16(false, ai, false, br, (short)0, aI,  false, false);
  }
  long co = (long)nloc;
#pragma unroll
  for (int v = 0; v < 8; ++v) {
    long b = row0 + v + ((lane >> 4) * 8);
    long idx = (b * 288 + p) * 64 + co;
    G2r[idx] = aRR[v] - aII[v];
    G2i[idx] = aI[v];
  }
}

// Inverse DFT along W with Hermitian fold (alpha_n) and 1/1156 scale.
__global__ void dft_w_inv_kernel(const float* __restrict__ G2r,
                                 const float* __restrict__ G2i,
                                 float* __restrict__ Htr, float* __restrict__ Hti) {
  __shared__ float cs[12 * 34], sn[12 * 34];
  int bm = blockIdx.x, b = bm / 24, m = bm % 24;
  int tid = threadIdx.x;
  for (int i = tid; i < 12 * 34; i += 256) {
    int n = i / 34, w = i % 34;
    float a = TWO_PI * (float)(n * w) / 34.0f;
    float s = ((n == 0) ? 1.0f : 2.0f) * (1.0f / 1156.0f);
    float sv, cv; __sincosf(a, &sv, &cv);
    cs[i] = cv * s; sn[i] = sv * s;
  }
  __syncthreads();
  long gbase = ((long)b * 288 + m * 12) * 64;
  long obase = ((long)b * 24 + m) * 34 * 64;
  for (int idx = tid; idx < 34 * 64; idx += 256) {
    int w = idx >> 6, c = idx & 63;
    float hr = 0.f, hi = 0.f;
    for (int n = 0; n < 12; ++n) {
      float gr = G2r[gbase + (long)n * 64 + c];
      float gi = G2i[gbase + (long)n * 64 + c];
      float cv = cs[n * 34 + w], sv = sn[n * 34 + w];
      hr += cv * gr - sv * gi;
      hi += cv * gi + sv * gr;
    }
    Htr[obase + idx] = hr;
    Hti[obase + idx] = hi;
  }
}

// Inverse DFT along H (Re part) + add pointwise conv + activation -> h
__global__ void combine_kernel(const float* __restrict__ Htr,
                               const float* __restrict__ Hti,
                               const float* __restrict__ x2,
                               float* __restrict__ h, int act) {
  __shared__ float cs[24], sn[24];
  int by = blockIdx.x, b = by / 34, y = by % 34;
  int tid = threadIdx.x;
  if (tid < 24) {
    int r = (tid < 12) ? tid : (tid + 10);
    float a = TWO_PI * (float)(r * y) / 34.0f;
    __sincosf(a, &sn[tid], &cs[tid]);
  }
  __syncthreads();
  long hbase = ((long)b * 34 + y) * 34 * 64;
  long tbase = (long)b * 24 * 34 * 64;
  for (int idx = tid; idx < 34 * 64; idx += 256) {
    float x1 = 0.f;
    for (int m = 0; m < 24; ++m) {
      long t = tbase + ((long)m * 34) * 64 + idx;
      x1 += cs[m] * Htr[t] - sn[m] * Hti[t];
    }
    h[hbase + idx] = apply_act(x1 + x2[hbase + idx], act);
  }
}

// pe1 8x8/stride4 conv (C->1) + gelu + pe2 scalar -> p49 [256,49]
__global__ void pe_kernel(const float* __restrict__ h,
                          const float* __restrict__ pe1w, const float* __restrict__ pe1b,
                          const float* __restrict__ pe2w, const float* __restrict__ pe2b,
                          float* __restrict__ p49) {
  int b = blockIdx.x, t = threadIdx.x;
  if (t >= 49) return;
  int i = t / 7, j = t % 7;
  const float* hb = h + (long)b * 34 * 34 * 64;
  float acc = 0.f;
  for (int dy = 0; dy < 8; ++dy)
    for (int dx = 0; dx < 8; ++dx) {
      const float* hp = hb + ((long)(4 * i + dy) * 34 + (4 * j + dx)) * 64;
      for (int c = 0; c < 64; ++c)
        acc += hp[c] * pe1w[(c * 8 + dy) * 8 + dx];
    }
  acc = apply_act(acc + pe1b[0], 2);
  p49[b * 49 + t] = acc * pe2w[0] + pe2b[0];
}

// ---------------------------------------------------------------------------
// Fused fc1(gelu)+fc2.  A = [h(crop 32x32, 64 ch) | emb] (65 cols).
// fc1 weights staged in LDS bf16; hidden 16x128 per wave staged in LDS;
// fc2 done on VALU (K=128, N=4).
// ---------------------------------------------------------------------------
__device__ inline v16bf load_a_hcat_fast(const float* __restrict__ hrow,
                                         float ev, int k0, int khalf) {
  v16bf f;
  if (k0 < 64) {                         // uniform branch: fully in h channels
    const float* p0 = hrow + k0 + khalf;
    const float* p1 = p0 + 16;
#pragma unroll
    for (int e = 0; e < 8; ++e) { f[e] = (__bf16)p0[e]; f[e + 8] = (__bf16)p1[e]; }
  } else {                               // k0 == 64: only k=64 (emb) non-zero
#pragma unroll
    for (int e = 0; e < 16; ++e) f[e] = (__bf16)0.0f;
    f[0] = (khalf == 0) ? (__bf16)ev : (__bf16)0.0f;  // lanes 0-15 carry k=64
  }
  return f;
}

__global__ void fc12_kernel(const float* __restrict__ h, const float* __restrict__ emb,
                            const float* __restrict__ fc1w, const float* __restrict__ fc1b,
                            const float* __restrict__ fc2w, const float* __restrict__ fc2b,
                            float* __restrict__ out) {
  __shared__ float hid[4][16][128];
  __shared__ __align__(32) __bf16 ldsW[128][112];  // [n][k], Kp=96 + 16 pad
  const int tid = threadIdx.x;

  for (int idx = tid; idx < 128 * 96; idx += 128) {
    int n = idx & 127, k = idx >> 7;
    float v = (k < 65) ? fc1w[k * 128 + n] : 0.0f;
    ldsW[n][(k & ~31) | (((k >> 3) & 1) << 4) | (((k >> 4) & 1) << 3) | (k & 7)] = (__bf16)v;
  }
  __syncthreads();

  const int lane = tid & 31, wave = tid >> 5;
  const long row0 = ((long)blockIdx.x * 4 + wave) * 16;
  long r = row0 + (lane & 15);
  long b = r >> 10, rem = r & 1023, y = rem >> 5, x = rem & 31;
  const float* hrow = h + ((b * 34 + y) * 34 + x) * 64;
  float ev = emb[b * 1156 + y * 34 + x];
  const int khalf = (lane >> 4) * 8;
  const int koff = (lane >> 4) * 16;

#pragma unroll
  for (int nt = 0; nt < 8; ++nt) {
    int col0 = nt * 16;
    v8f acc = {};
#pragma unroll
    for (int k0 = 0; k0 < 96; k0 += 32) {
      v16bf a = load_a_hcat_fast(hrow, ev, k0, khalf);
      v16bf bfrag = *(const v16bf*)&ldsW[col0 + (lane & 15)][k0 + koff];
      acc = __builtin_amdgcn_wmma_f32_16x16x32_bf16(false, a, false, bfrag, (short)0,
                                                    acc, false, false);
    }
    int n = col0 + (lane & 15);
    float bb = fc1b[n];
#pragma unroll
    for (int v = 0; v < 8; ++v) {
      int m = v + ((lane >> 4) * 8);
      float xx = acc[v] + bb;
      hid[wave][m][n] = 0.5f * xx * (1.0f + erff(xx * 0.70710678118f));
    }
  }
  __syncthreads();
#pragma unroll
  for (int t = 0; t < 2; ++t) {
    int o = lane * 2 + t;                // 64 outputs per wave (16 px * 4)
    int px = o >> 2, j = o & 3;
    float acc = fc2b[j];
    for (int k = 0; k < 128; ++k) acc += hid[wave][px][k] * fc2w[k * 4 + j];
    out[(row0 + px) * 4 + j] = acc;
  }
}

// ---------------------------------------------------------------------------
// Host-side orchestration
// ---------------------------------------------------------------------------
static inline void launch_gemm(const float* A, long lda, const float* B, long brs,
                               long bcs, const float* bias, float* C, long ldc,
                               long M, long N, int K, int act, hipStream_t s) {
  int Kp = (K + 31) & ~31;
  size_t shmem = (size_t)64 * (size_t)(Kp + 16) * 2;   // bf16 B panel
  dim3 g((unsigned)((M + 15) / 16), (unsigned)((N + 63) / 64));
  gemm_wmma_kernel<<<g, 128, shmem, s>>>(A, lda, B, brs, bcs, bias, C, ldc,
                                         M, N, K, act);
}

extern "C" void kernel_launch(void* const* d_in, const int* in_sizes, int n_in,
                              void* d_out, int out_size, void* d_ws, size_t ws_size,
                              hipStream_t stream) {
  const float* x    = (const float*)d_in[0];
  const float* grd  = (const float*)d_in[1];
  const float* se   = (const float*)d_in[2];
  const float* fc0w = (const float*)d_in[3];
  const float* fc0b = (const float*)d_in[4];
  const float* scw1 = (const float*)d_in[5];
  const float* scw2 = (const float*)d_in[6];
  const float* wcw  = (const float*)d_in[7];
  const float* wcb  = (const float*)d_in[8];
  const float* pe1w = (const float*)d_in[9];
  const float* pe1b = (const float*)d_in[10];
  const float* pe2w = (const float*)d_in[11];
  const float* pe2b = (const float*)d_in[12];
  const float* spw1 = (const float*)d_in[13];
  const float* spb1 = (const float*)d_in[14];
  const float* spw2 = (const float*)d_in[15];
  const float* spb2 = (const float*)d_in[16];
  const float* spw3 = (const float*)d_in[17];
  const float* spb3 = (const float*)d_in[18];
  const float* xpw1 = (const float*)d_in[19];
  const float* xpb1 = (const float*)d_in[20];
  const float* xpw2 = (const float*)d_in[21];
  const float* xpb2 = (const float*)d_in[22];
  const float* xpw3 = (const float*)d_in[23];
  const float* xpb3 = (const float*)d_in[24];
  const float* puw1 = (const float*)d_in[25];
  const float* pub1 = (const float*)d_in[26];
  const float* puw2 = (const float*)d_in[27];
  const float* pub2 = (const float*)d_in[28];
  const float* puw3 = (const float*)d_in[29];
  const float* pub3 = (const float*)d_in[30];
  const float* fc1w = (const float*)d_in[31];
  const float* fc1b = (const float*)d_in[32];
  const float* fc2w = (const float*)d_in[33];
  const float* fc2b = (const float*)d_in[34];
  float* out = (float*)d_out;
  float* ws  = (float*)d_ws;

  const long B = 256;
  const long HS = B * 34 * 34 * 64;
  const long HM = B * 24 * 34 * 64;
  const long GM = B * 24 * 12 * 64;

  float* h   = ws;
  float* h2  = h + HS;
  float* Htr = h2 + HS;
  float* Hti = Htr + HM;
  float* Gr  = Hti + HM;
  float* Gi  = Gr + GM;
  float* G2r = Gi + GM;
  float* G2i = G2r + GM;
  float* sm  = G2i + GM;
  float* s1 = sm;           float* s2 = s1 + 256 * 32;
  float* embin = s2 + 256 * 32;
  float* p49 = embin + 256 * 32;
  float* e1 = p49 + 256 * 49;  float* e2 = e1 + 256 * 32;
  float* pu1b = e2 + 256 * 32; float* pu2b = pu1b + 256 * 128;
  float* embf = pu2b + 256 * 256;
  float* Acat = Htr;            // alias (used before DFT loop)
  float* htmp = Gr;             // alias (used before DFT loop)

  // --- sentence MLP -> embin cols 16..31 ---
  launch_gemm(se, 384, spw1, 32, 1, spb1, s1, 32, 256, 32, 384, 1, stream);
  launch_gemm(s1, 32, spw2, 32, 1, spb2, s2, 32, 256, 32, 32, 1, stream);
  launch_gemm(s2, 32, spw3, 16, 1, spb3, embin + 16, 32, 256, 16, 32, 0, stream);

  // --- fc0 then zero-pad to 34x34 ---
  {
    long tot = 262144L * 42;
    concat42_kernel<<<(unsigned)((tot + 255) / 256), 256, 0, stream>>>(x, grd, Acat, tot);
    launch_gemm(Acat, 42, fc0w, 64, 1, fc0b, htmp, 64, 262144, 64, 42, 0, stream);
    pad_kernel<<<(unsigned)((HS + 255) / 256), 256, 0, stream>>>(htmp, h, HS);
  }

  // --- 4 FNO layers ---
  for (int i = 0; i < 4; ++i) {
    const float* w1 = scw1 + (long)i * 64 * 64 * 144 * 2;
    const float* w2 = scw2 + (long)i * 64 * 64 * 144 * 2;
    dft_h_fwd_kernel<<<(unsigned)(B * 24), 256, 0, stream>>>(h, Htr, Hti);
    dft_w_fwd_kernel<<<(unsigned)(B * 24), 256, 0, stream>>>(Htr, Hti, Gr, Gi);
    mode_mix_kernel<<<dim3(16, 288), 128, 0, stream>>>(Gr, Gi, w1, w2, G2r, G2i);
    dft_w_inv_kernel<<<(unsigned)(B * 24), 256, 0, stream>>>(G2r, G2i, Htr, Hti);
    launch_gemm(h, 64, wcw + (long)i * 64 * 64, 1, 64, wcb + (long)i * 64,
                h2, 64, B * 34 * 34, 64, 64, 0, stream);
    combine_kernel<<<(unsigned)(B * 34), 256, 0, stream>>>(Htr, Hti, h2, h,
                                                           (i < 3) ? 2 : 0);
  }

  // --- patch embed + xp MLP -> embin cols 0..15 ---
  pe_kernel<<<(unsigned)B, 64, 0, stream>>>(h, pe1w, pe1b, pe2w, pe2b, p49);
  launch_gemm(p49, 49, xpw1, 32, 1, xpb1, e1, 32, 256, 32, 49, 3, stream);
  launch_gemm(e1, 32, xpw2, 32, 1, xpb2, e2, 32, 256, 32, 32, 3, stream);
  launch_gemm(e2, 32, xpw3, 16, 1, xpb3, embin, 32, 256, 16, 32, 0, stream);

  // --- pu MLP -> emb plane [256, 34*34] ---
  launch_gemm(embin, 32, puw1, 128, 1, pub1, pu1b, 128, 256, 128, 32, 3, stream);
  launch_gemm(pu1b, 128, puw2, 256, 1, pub2, pu2b, 256, 256, 256, 128, 3, stream);
  launch_gemm(pu2b, 256, puw3, 1156, 1, pub3, embf, 1156, 256, 1156, 256, 0, stream);

  // --- fused fc1(gelu)+fc2 ---
  fc12_kernel<<<4096, 128, 0, stream>>>(h, embf, fc1w, fc1b, fc2w, fc2b, out);
}
